// EMASplitModel_35588099014904
// MI455X (gfx1250) — matrix operand order repaired and checked by reading, not compiled, not run on hardware
//
#include <hip/hip_runtime.h>
#include <math.h>

// ---------------------------------------------------------------------------
// EMASplitModel for MI455X (gfx1250, wave32, WMMA).
//
// Algorithmic move: the sequential 256x256 rank-1 EMA scan is linear in M with
// M0 = 0, and only M_final @ q is needed. So we run a BACKWARD vector scan:
//   z_T = q;  z_t = z_{t+1} - 0.05*(kn_t . z_{t+1})*kn_t;
//   c   = sum_t 0.05 * ks_t * (kn_t . z_{t+1})
// exact, ~170x less work than materializing M.
//
// All large GEMMs run through v_wmma_f32_16x16x32_bf16. The tiled GEMM uses
// double-buffered LDS fed by GLOBAL_LOAD_ASYNC_TO_LDS_B128 (ASYNCcnt) when the
// toolchain exposes the builtins, overlapping HBM traffic with WMMA issue.
// ---------------------------------------------------------------------------

#define BDIM 16
#define LDIM 1024
#define HDIM 512
#define VDIM 32000
#define HALFD 256

typedef __bf16 v16bf __attribute__((ext_vector_type(16)));
typedef float  v8f   __attribute__((ext_vector_type(8)));
typedef int    v4i   __attribute__((ext_vector_type(4)));

#if __has_builtin(__builtin_amdgcn_global_load_async_to_lds_b128) && \
    __has_builtin(__builtin_amdgcn_s_wait_asynccnt)
#define HAVE_ASYNC_LDS 1
#else
#define HAVE_ASYNC_LDS 0
#endif

#if HAVE_ASYNC_LDS
typedef __attribute__((address_space(1))) v4i* as1_v4i_p;  // global int4*
typedef __attribute__((address_space(3))) v4i* as3_v4i_p;  // LDS    int4*
__device__ __forceinline__ void async_cp16(const void* g, void* l) {
  // One wave-wide instruction: 16 bytes per lane, global -> LDS, ASYNCcnt++.
  __builtin_amdgcn_global_load_async_to_lds_b128(
      (as1_v4i_p)g, (as3_v4i_p)l, 0, 0);
}
#endif

__device__ __forceinline__ v16bf ld2x16(const __bf16* p0, const __bf16* p1) {
  union { uint4 q[2]; v16bf v; } u;
  u.q[0] = *reinterpret_cast<const uint4*>(p0);
  u.q[1] = *reinterpret_cast<const uint4*>(p1);
  return u.v;
}

// One BK=32 step of WMMA work for a wave: 4x2 fragments of 16x16.
__device__ __forceinline__ void compute_step(const __bf16* sA, const __bf16* sB,
                                             v8f (&acc)[4][2],
                                             int wr, int wc, int mRow, int kh)
{
  v16bf bfrag[2];
#pragma unroll
  for (int fc = 0; fc < 2; ++fc) {
    // B (32x16): lane n holds col n, K = {kh*8..+7, 16+kh*8..+7}.
    const __bf16* p = &sB[(wc + fc * 16 + mRow) * 32 + kh * 8];
    bfrag[fc] = ld2x16(p, p + 16);
  }
#pragma unroll
  for (int fr = 0; fr < 4; ++fr) {
    // A (16x32): lane holds row mRow, K = {kh*8..+7, 16+kh*8..+7}.
    const __bf16* p = &sA[(wr + fr * 16 + mRow) * 32 + kh * 8];
    v16bf afrag = ld2x16(p, p + 16);
#pragma unroll
    for (int fc = 0; fc < 2; ++fc)
      acc[fr][fc] = __builtin_amdgcn_wmma_f32_16x16x32_bf16(
          false, afrag, false, bfrag[fc], (short)0, acc[fr][fc], false, false);
  }
}

// ---------------------------------------------------------------------------
// Tiled WMMA GEMM: C[M,N] = A[M,K](bf16) @ W[N,K]^T(bf16)  (+bias)
// Block: 256 threads (8 waves), tile 128x128, BK=32, double-buffered LDS.
// MODE 0: f32 out (+bias). MODE 2: relu -> bf16 out (+bias).
// MODE 3: scatter f32 to ks_all/ke_all in [L,B,HALF] time-major layout.
// ---------------------------------------------------------------------------
template<int MODE>
__global__ __launch_bounds__(256)
void gemm_tile(const __bf16* __restrict__ A, const __bf16* __restrict__ W,
               const float* __restrict__ bias,
               float* __restrict__ Cf, __bf16* __restrict__ Cb,
               float* __restrict__ ks, float* __restrict__ ke,
               int M, int N, int K)
{
  __shared__ __align__(16) __bf16 sA[2][128 * 32];
  __shared__ __align__(16) __bf16 sB[2][128 * 32];

  const int tid   = threadIdx.x;
  const int tileN = blockIdx.x * 128;
  const int tileM = blockIdx.y * 128;
  const int wave  = tid >> 5;
  const int lane  = tid & 31;
  const int wr    = (wave >> 2) * 64;   // wave row offset in tile
  const int wc    = (wave & 3) * 32;    // wave col offset in tile
  const int mRow  = lane & 15;
  const int kh    = lane >> 4;          // K-half select per ISA layout

  // Per-thread staging coordinates: 2 x 16B chunks of A and of W per step.
  const int r0 = tid >> 2,          s0 = (tid & 3) * 8;
  const int r1 = (tid + 256) >> 2,  s1 = s0;           // (e&3) identical

  v8f acc[4][2];
  const v8f vz = {0.f, 0.f, 0.f, 0.f, 0.f, 0.f, 0.f, 0.f};
  for (int i = 0; i < 4; ++i)
    for (int j = 0; j < 2; ++j) acc[i][j] = vz;

#if HAVE_ASYNC_LDS
  // ---- async double-buffered pipeline: ASYNCcnt-tracked copies ----
  auto stage = [&](int buf, int k0) {
    async_cp16(&A[(size_t)(tileM + r0) * K + k0 + s0], &sA[buf][r0 * 32 + s0]);
    async_cp16(&W[(size_t)(tileN + r0) * K + k0 + s0], &sB[buf][r0 * 32 + s0]);
    async_cp16(&A[(size_t)(tileM + r1) * K + k0 + s1], &sA[buf][r1 * 32 + s1]);
    async_cp16(&W[(size_t)(tileN + r1) * K + k0 + s1], &sB[buf][r1 * 32 + s1]);
  };
  stage(0, 0);
  int buf = 0;
  for (int k0 = 0; k0 < K; k0 += 32, buf ^= 1) {
    if (k0 + 32 < K) {
      stage(buf ^ 1, k0 + 32);                 // prefetch next tile into LDS
      __builtin_amdgcn_s_wait_asynccnt(4);     // this wave's current tile done
    } else {
      __builtin_amdgcn_s_wait_asynccnt(0);
    }
    __syncthreads();                           // all waves' copies visible
    compute_step(&sA[buf][0], &sB[buf][0], acc, wr, wc, mRow, kh);
    __syncthreads();                           // done reading before overwrite
  }
#else
  // ---- fallback: synchronous VGPR staging, single logical buffer ----
  for (int k0 = 0; k0 < K; k0 += 32) {
    __syncthreads();
    *reinterpret_cast<uint4*>(&sA[0][r0 * 32 + s0]) =
        *reinterpret_cast<const uint4*>(&A[(size_t)(tileM + r0) * K + k0 + s0]);
    *reinterpret_cast<uint4*>(&sB[0][r0 * 32 + s0]) =
        *reinterpret_cast<const uint4*>(&W[(size_t)(tileN + r0) * K + k0 + s0]);
    *reinterpret_cast<uint4*>(&sA[0][r1 * 32 + s1]) =
        *reinterpret_cast<const uint4*>(&A[(size_t)(tileM + r1) * K + k0 + s1]);
    *reinterpret_cast<uint4*>(&sB[0][r1 * 32 + s1]) =
        *reinterpret_cast<const uint4*>(&W[(size_t)(tileN + r1) * K + k0 + s1]);
    __syncthreads();
    compute_step(&sA[0][0], &sB[0][0], acc, wr, wc, mRow, kh);
  }
#endif

  // Epilogue: C/D layout -> VGPR r: lanes 0-15 M=r, lanes 16-31 M=8+r; N=lane%16.
#pragma unroll
  for (int fr = 0; fr < 4; ++fr) {
#pragma unroll
    for (int fc = 0; fc < 2; ++fc) {
      int col = tileN + wc + fc * 16 + mRow;
      float bv = (MODE != 3 && bias) ? bias[col] : 0.f;
#pragma unroll
      for (int r = 0; r < 8; ++r) {
        int row = tileM + wr + fr * 16 + kh * 8 + r;
        float v = acc[fr][fc][r] + bv;
        if (MODE == 0) {
          Cf[(size_t)row * N + col] = v;
        } else if (MODE == 2) {
          Cb[(size_t)row * N + col] = (__bf16)fmaxf(v, 0.f);
        } else {  // MODE 3: row = b*L + l  ->  [l*B + b] time-major rows
          int b = row >> 10, l = row & (LDIM - 1);
          size_t o = (size_t)(l * BDIM + b) * HALFD;
          if (col < HALFD) ks[o + col] = v;
          else             ke[o + col - HALFD] = v;
        }
      }
    }
  }
}

// ---------------------------------------------------------------------------
// Small-M GEMM: C[16,N] = A[16,K](bf16) @ W[N,K]^T(bf16) + bias. One wave per
// 16x16 output tile, K-loop of WMMAs, fragments loaded straight from global.
// ---------------------------------------------------------------------------
__global__ __launch_bounds__(32)
void gemm_m16(const __bf16* __restrict__ A, const __bf16* __restrict__ W,
              const float* __restrict__ bias, float* __restrict__ C,
              int N, int K)
{
  const int lane = threadIdx.x;
  const int mRow = lane & 15, kh = lane >> 4;
  const int n0 = blockIdx.x * 16;
  v8f acc = {0.f, 0.f, 0.f, 0.f, 0.f, 0.f, 0.f, 0.f};
  for (int k0 = 0; k0 < K; k0 += 32) {
    const __bf16* pa = &A[(size_t)mRow * K + k0 + kh * 8];
    v16bf af = ld2x16(pa, pa + 16);
    const __bf16* pb = &W[(size_t)(n0 + mRow) * K + k0 + kh * 8];
    v16bf bf = ld2x16(pb, pb + 16);
    acc = __builtin_amdgcn_wmma_f32_16x16x32_bf16(
        false, af, false, bf, (short)0, acc, false, false);
  }
  int col = n0 + mRow;
  float bv = bias ? bias[col] : 0.f;
#pragma unroll
  for (int r = 0; r < 8; ++r)
    C[(size_t)(kh * 8 + r) * N + col] = acc[r] + bv;
}

// ---------------------------------------------------------------------------
// Embedding gather: h(f32) and hbf(bf16), rows ordered b*L+l.
// ---------------------------------------------------------------------------
__global__ __launch_bounds__(256)
void embed_gather(const int* __restrict__ seq, const float* __restrict__ emb,
                  float* __restrict__ h, __bf16* __restrict__ hbf)
{
  int idx = blockIdx.x * 256 + threadIdx.x;      // < B*L*H, exact grid
  int row = idx >> 9, col = idx & (HDIM - 1);
  int tok = seq[row];
  float v = emb[(size_t)tok * HDIM + col];
  h[idx] = v;
  hbf[idx] = (__bf16)v;
}

// ---------------------------------------------------------------------------
// Residual + LayerNorm, emits bf16 for the key-projection GEMM.
// ---------------------------------------------------------------------------
__global__ __launch_bounds__(256)
void ln_res(const float* __restrict__ h, const float* __restrict__ ff,
            const float* __restrict__ gamma, const float* __restrict__ beta,
            __bf16* __restrict__ out)
{
  int m = blockIdx.x, t = threadIdx.x;
  size_t base = (size_t)m * HDIM;
  float x0 = h[base + t] + ff[base + t];
  float x1 = h[base + t + 256] + ff[base + t + 256];
  __shared__ float s1[256], s2[256];
  s1[t] = x0 + x1;
  s2[t] = x0 * x0 + x1 * x1;
  __syncthreads();
  for (int off = 128; off > 0; off >>= 1) {
    if (t < off) { s1[t] += s1[t + off]; s2[t] += s2[t + off]; }
    __syncthreads();
  }
  float mu  = s1[0] * (1.f / HDIM);
  float var = s2[0] * (1.f / HDIM) - mu * mu;
  float rs  = rsqrtf(var + 1e-5f);
  out[base + t]       = (__bf16)((x0 - mu) * rs * gamma[t] + beta[t]);
  out[base + t + 256] = (__bf16)((x1 - mu) * rs * gamma[t + 256] + beta[t + 256]);
}

// ---------------------------------------------------------------------------
// Per-row inverse L2 norms of the time-major key arrays.
// ---------------------------------------------------------------------------
__global__ __launch_bounds__(256)
void key_norms(const float* __restrict__ ks, const float* __restrict__ ke,
               float* __restrict__ rns, float* __restrict__ rne)
{
  int row = blockIdx.x, t = threadIdx.x;
  size_t base = (size_t)row * HALFD;
  float a = ks[base + t], b = ke[base + t];
  __shared__ float s1[256], s2[256];
  s1[t] = a * a;
  s2[t] = b * b;
  __syncthreads();
  for (int off = 128; off > 0; off >>= 1) {
    if (t < off) { s1[t] += s1[t + off]; s2[t] += s2[t + off]; }
    __syncthreads();
  }
  if (t == 0) {
    rns[row] = 1.f / fmaxf(sqrtf(s1[0]), 1e-12f);
    rne[row] = 1.f / fmaxf(sqrtf(s2[0]), 1e-12f);
  }
}

// ---------------------------------------------------------------------------
// Backward EMA vector scan. 32 chains = 32 single-wave blocks.
// z in 8 VGPRs/lane; dot reduced with 5 shfl_xor steps (wave32).
// ---------------------------------------------------------------------------
__global__ __launch_bounds__(32)
void ema_scan(const float* __restrict__ ksall, const float* __restrict__ keall,
              const float* __restrict__ rns, const float* __restrict__ rne,
              float* __restrict__ cvec)
{
  const int b     = blockIdx.x & 15;
  const int which = blockIdx.x >> 4;
  const float* kall = which ? keall : ksall;
  const float* rn   = which ? rne   : rns;
  const int lane = threadIdx.x;

  float z[8], c[8], k[8];
  {
    const float4* q = reinterpret_cast<const float4*>(
        kall + (size_t)((LDIM - 1) * BDIM + b) * HALFD + lane * 8);
    float4 q0 = q[0], q1 = q[1];
    z[0]=q0.x; z[1]=q0.y; z[2]=q0.z; z[3]=q0.w;
    z[4]=q1.x; z[5]=q1.y; z[6]=q1.z; z[7]=q1.w;
  }
#pragma unroll
  for (int i = 0; i < 8; ++i) c[i] = 0.f;

  for (int t = LDIM - 2; t >= 0; --t) {
    const float* krow = kall + (size_t)(t * BDIM + b) * HALFD;
    if (t > 0) __builtin_prefetch(krow - BDIM * HALFD, 0, 0);
    const float4* kp = reinterpret_cast<const float4*>(krow + lane * 8);
    float4 k0 = kp[0], k1 = kp[1];
    k[0]=k0.x; k[1]=k0.y; k[2]=k0.z; k[3]=k0.w;
    k[4]=k1.x; k[5]=k1.y; k[6]=k1.z; k[7]=k1.w;
    float r = rn[t * BDIM + b];
    float p = 0.f;
#pragma unroll
    for (int i = 0; i < 8; ++i) p += k[i] * z[i];
    p *= r;                                   // dot(kn, z)
    for (int off = 16; off > 0; off >>= 1) p += __shfl_xor(p, off, 32);
    float coef = 0.05f * p;                   // (1 - ALPHA) * d
    float cr = coef * r;
#pragma unroll
    for (int i = 0; i < 8; ++i) {
      c[i] += coef * k[i];                    // c += 0.05*d*ks
      z[i] -= cr * k[i];                      // z -= 0.05*d*kn
    }
  }
  float* out = cvec + (size_t)b * HDIM + which * HALFD + lane * 8;
#pragma unroll
  for (int i = 0; i < 8; ++i) out[i] = c[i];
}

__global__ __launch_bounds__(256)
void cvt_f32_bf16(const float* __restrict__ s, __bf16* __restrict__ d, int n)
{
  int i = blockIdx.x * 256 + threadIdx.x;
  if (i < n) d[i] = (__bf16)s[i];
}

// ---------------------------------------------------------------------------
extern "C" void kernel_launch(void* const* d_in, const int* in_sizes, int n_in,
                              void* d_out, int out_size, void* d_ws, size_t ws_size,
                              hipStream_t stream)
{
  (void)in_sizes; (void)n_in; (void)out_size; (void)ws_size;
  const int*   seq   = (const int*)  d_in[0];
  const float* emb   = (const float*)d_in[1];
  const float* w1    = (const float*)d_in[2];
  const float* b1    = (const float*)d_in[3];
  const float* w2    = (const float*)d_in[4];
  const float* b2    = (const float*)d_in[5];
  const float* gamma = (const float*)d_in[6];
  const float* beta  = (const float*)d_in[7];
  const float* semw  = (const float*)d_in[8];
  const float* epiw  = (const float*)d_in[9];
  const float* rpw   = (const float*)d_in[10];
  const float* rpbi  = (const float*)d_in[11];
  const float* outw  = (const float*)d_in[12];
  const float* outb  = (const float*)d_in[13];
  float* logits = (float*)d_out;

  // Bump-allocate workspace (~205 MB total).
  char* base = (char*)d_ws;
  size_t off = 0;
  auto alloc = [&](size_t bytes) -> char* {
    char* p = base + off;
    off = (off + bytes + 255) & ~(size_t)255;
    return p;
  };
  const size_t ML = (size_t)BDIM * LDIM;                 // 16384 rows
  float*  h    = (float*) alloc(ML * HDIM * 4);
  __bf16* hbf  = (__bf16*)alloc(ML * HDIM * 2);
  __bf16* c1   = (__bf16*)alloc(ML * 2 * HDIM * 2);
  float*  c2   = (float*) alloc(ML * HDIM * 4);
  __bf16* hnb  = (__bf16*)alloc(ML * HDIM * 2);
  float*  ksA  = (float*) alloc(ML * HALFD * 4);
  float*  keA  = (float*) alloc(ML * HALFD * 4);
  float*  rns  = (float*) alloc(ML * 4);
  float*  rne  = (float*) alloc(ML * 4);
  __bf16* w1b  = (__bf16*)alloc((size_t)2 * HDIM * HDIM * 2);
  __bf16* w2b  = (__bf16*)alloc((size_t)HDIM * 2 * HDIM * 2);
  __bf16* kwb  = (__bf16*)alloc((size_t)HDIM * HDIM * 2);   // [sem;epi]
  __bf16* rwb  = (__bf16*)alloc((size_t)HDIM * HDIM * 2);
  __bf16* owb  = (__bf16*)alloc((size_t)VDIM * HDIM * 2);
  float*  cvec = (float*) alloc((size_t)BDIM * HDIM * 4);
  __bf16* cvb  = (__bf16*)alloc((size_t)BDIM * HDIM * 2);
  float*  rvec = (float*) alloc((size_t)BDIM * HDIM * 4);
  __bf16* rvb  = (__bf16*)alloc((size_t)BDIM * HDIM * 2);

  auto cvt = [&](const float* s, __bf16* d, size_t n) {
    cvt_f32_bf16<<<dim3((unsigned)((n + 255) / 256)), 256, 0, stream>>>(s, d, (int)n);
  };
  // Weight conversions to bf16 (kproj = concat(sem_w, epi_w)).
  cvt(w1,  w1b, (size_t)2 * HDIM * HDIM);
  cvt(w2,  w2b, (size_t)HDIM * 2 * HDIM);
  cvt(semw, kwb,                         (size_t)HALFD * HDIM);
  cvt(epiw, kwb + (size_t)HALFD * HDIM,  (size_t)HALFD * HDIM);
  cvt(rpw, rwb, (size_t)HDIM * HDIM);
  cvt(outw, owb, (size_t)VDIM * HDIM);

  // 1. Embedding gather.
  embed_gather<<<dim3((unsigned)(ML * HDIM / 256)), 256, 0, stream>>>(seq, emb, h, hbf);

  // 2. FF1: relu(h @ w1^T + b1) -> bf16.
  gemm_tile<2><<<dim3((2 * HDIM) / 128, (unsigned)(ML / 128)), 256, 0, stream>>>(
      hbf, w1b, b1, nullptr, c1, nullptr, nullptr, (int)ML, 2 * HDIM, HDIM);

  // 3. FF2: c1 @ w2^T + b2 -> f32.
  gemm_tile<0><<<dim3(HDIM / 128, (unsigned)(ML / 128)), 256, 0, stream>>>(
      c1, w2b, b2, c2, nullptr, nullptr, nullptr, (int)ML, HDIM, 2 * HDIM);

  // 4. Residual + LayerNorm -> bf16.
  ln_res<<<dim3((unsigned)ML), 256, 0, stream>>>(h, c2, gamma, beta, hnb);

  // 5. Key projections, scattered time-major into ks_all / ke_all.
  gemm_tile<3><<<dim3(HDIM / 128, (unsigned)(ML / 128)), 256, 0, stream>>>(
      hnb, kwb, nullptr, nullptr, nullptr, ksA, keA, (int)ML, HDIM, HDIM);

  // 6. Inverse norms, then backward vector scan (replaces rank-1 matrix scan).
  key_norms<<<dim3((unsigned)ML), 256, 0, stream>>>(ksA, keA, rns, rne);
  ema_scan<<<dim3(32), 32, 0, stream>>>(ksA, keA, rns, rne, cvec);

  // 7. Head: r = c @ rp_w^T + rp_b ; logits = r @ out_w^T + out_b.
  cvt(cvec, cvb, (size_t)BDIM * HDIM);
  gemm_m16<<<dim3(HDIM / 16), 32, 0, stream>>>(cvb, rwb, rpbi, rvec, HDIM, HDIM);
  cvt(rvec, rvb, (size_t)BDIM * HDIM);
  gemm_m16<<<dim3(VDIM / 16), 32, 0, stream>>>(rvb, owb, outb, logits, VDIM, HDIM);
}
